// HCSMoEQwen3MoeSparseMoeBlock_11536282157453
// MI455X (gfx1250) — matrix operand order repaired and checked by reading, not compiled, not run on hardware
//
#include <hip/hip_runtime.h>
#include <hip/hip_bf16.h>

// ---------------------------------------------------------------------------
// HCSMoE Qwen3 sparse-MoE block for MI455X (gfx1250, wave32, WMMA).
//   T=2048 tokens, H=2048, I=768, E=32 experts, G=8 groups, TOP_K=8.
// Pipeline per call:
//   1) x fp32 -> f16 (ws)
//   2) router: per-token wave32, lane==expert; softmax + top-8 + group-collapse
//   3) per group g:
//        a) gather+convert dominant expert's gate_up / down weights to f16
//        b) GEMM1 (fused): act = silu(x*gu_g^T) * (x*gu_u^T)   [f16 out]
//        c) GEMM2: out (+)= w_tg[:,g] * (act * dn^T)           [fused epilogue]
// GEMM k-loop: double-buffered LDS, ALL tile staging via
// GLOBAL_LOAD_ASYNC_TO_LDS_B128 (ASYNCcnt), V_WMMA_F32_16X16X32_F16 compute.
// ---------------------------------------------------------------------------

typedef __attribute__((ext_vector_type(16))) _Float16 v16h;
typedef __attribute__((ext_vector_type(8)))  _Float16 v8h;
typedef __attribute__((ext_vector_type(8)))  float    v8f;
typedef int v4i __attribute__((__vector_size__(4 * sizeof(int))));

#define T_TOK 2048
#define H_DIM 2048
#define I_DIM 768
#define G_NUM 8

#if __has_builtin(__builtin_amdgcn_global_load_async_to_lds_b128) && \
    __has_builtin(__builtin_amdgcn_s_wait_asynccnt)
#define USE_ASYNC_LDS 1
#endif

#ifdef USE_ASYNC_LDS
#define WAIT_ASYNC(n) __builtin_amdgcn_s_wait_asynccnt(n)
#else
#define WAIT_ASYNC(n) ((void)0)
#endif

__device__ __forceinline__ void copy16_g2l(const _Float16* gsrc,
                                           _Float16* ldst) {
#ifdef USE_ASYNC_LDS
  __builtin_amdgcn_global_load_async_to_lds_b128(
      (__attribute__((address_space(1))) v4i*)gsrc,
      (__attribute__((address_space(3))) v4i*)ldst, 0, 0);
#else
  *(float4*)ldst = *(const float4*)gsrc;
#endif
}

// ------------------------------- x -> f16 ----------------------------------
__global__ void convert_x_kernel(const float* __restrict__ x,
                                 _Float16* __restrict__ xh) {
  long long i = ((long long)blockIdx.x * 256 + threadIdx.x) * 4;
  float4 v = *(const float4*)(x + i);
  union { _Float16 h[4]; float2 f; } u;
  u.h[0] = (_Float16)v.x; u.h[1] = (_Float16)v.y;
  u.h[2] = (_Float16)v.z; u.h[3] = (_Float16)v.w;
  *(float2*)(xh + i) = u.f;
}

// ---------------------- gather+convert expert weights ----------------------
// dst[i] = (f16) srcBase[dom[g]*stride + i], 8 elements per thread.
__global__ void convert_w_kernel(const float* __restrict__ srcBase,
                                 long long stride,
                                 const int* __restrict__ dom, int g,
                                 _Float16* __restrict__ dst) {
  long long i = ((long long)blockIdx.x * 256 + threadIdx.x) * 8;
  const float* s = srcBase + (long long)dom[g] * stride + i;
  float4 f0 = *(const float4*)s;
  float4 f1 = *(const float4*)(s + 4);
  union { _Float16 h[8]; float4 v; } u;
  u.h[0] = (_Float16)f0.x; u.h[1] = (_Float16)f0.y;
  u.h[2] = (_Float16)f0.z; u.h[3] = (_Float16)f0.w;
  u.h[4] = (_Float16)f1.x; u.h[5] = (_Float16)f1.y;
  u.h[6] = (_Float16)f1.z; u.h[7] = (_Float16)f1.w;
  *(float4*)(dst + i) = u.v;
}

// ------------------------------- router ------------------------------------
// One wave32 per token; lane e owns expert e (E == 32 == wave32).
__global__ void router_kernel(const float* __restrict__ x,
                              const float* __restrict__ gw,
                              const int*   __restrict__ merge_groups,
                              float*       __restrict__ wtg) {
  const int lane = threadIdx.x & 31;
  const int wave = threadIdx.x >> 5;
  const int t = blockIdx.x * 8 + wave;

  const float* xr = x  + (long long)t    * H_DIM;
  const float* gr = gw + (long long)lane * H_DIM;

  float acc = 0.f;
  for (int k = 0; k < H_DIM; k += 4) {
    float4 xv = *(const float4*)(xr + k);
    float4 gv = *(const float4*)(gr + k);
    acc = fmaf(xv.x, gv.x, acc);
    acc = fmaf(xv.y, gv.y, acc);
    acc = fmaf(xv.z, gv.z, acc);
    acc = fmaf(xv.w, gv.w, acc);
  }

  // softmax over 32 experts (one per lane)
  float m = acc;
  for (int off = 16; off; off >>= 1) m = fmaxf(m, __shfl_xor(m, off));
  float p = __expf(acc - m);
  float s = p;
  for (int off = 16; off; off >>= 1) s += __shfl_xor(s, off);
  float prob = p / s;

  // iterative top-8 across lanes
  float rem = prob;
  bool  sel = false;
  float ksum = 0.f;
  for (int k = 0; k < 8; ++k) {
    float mv = rem;
    for (int off = 16; off; off >>= 1) mv = fmaxf(mv, __shfl_xor(mv, off));
    unsigned long long b = __ballot(rem == mv);
    int win = __ffsll((unsigned long long)b) - 1;
    if (lane == win) { sel = true; rem = -1.f; }
    ksum += mv;
  }
  float wn  = sel ? (prob / ksum) : 0.f;
  int   grp = merge_groups[lane];

  for (int g = 0; g < G_NUM; ++g) {
    float c = (sel && grp == g) ? wn : 0.f;
    for (int off = 16; off; off >>= 1) c += __shfl_xor(c, off);
    if (lane == 0) wtg[(long long)t * G_NUM + g] = c;
  }
}

// ------------------------------- WMMA GEMM ---------------------------------
// BM=BN=128, BK=32; 8 waves (2x4), wave tile 64x32 per output matrix.
// MODE 0 (gate_up fused): two B tiles (gate rows n, up rows I_DIM+n);
//   epilogue: actOut[t,n] = (f16)( silu(accG) * accU ),  N = I_DIM.
// MODE 1 (down):          epilogue: C = (accum?C:0) + wtg[t*8+g]*acc.
// All A/B tile staging: async global->LDS b128 copies (ASYNCcnt).
#define LDS_STRIDE 40   // f16 per LDS row (20 dwords -> conflict-free b128)

template <int MODE>
__global__ __launch_bounds__(256)
void gemm_wmma_kernel(const _Float16* __restrict__ A,
                      const _Float16* __restrict__ B0,
                      const _Float16* __restrict__ B1,
                      _Float16* __restrict__ actOut,
                      float* __restrict__ C,
                      const float* __restrict__ wtg,
                      int N, int K, int g, int accum) {
  __shared__ __align__(16) _Float16 lA [2][128 * LDS_STRIDE];
  __shared__ __align__(16) _Float16 lB0[2][128 * LDS_STRIDE];
  __shared__ __align__(16) _Float16 lB1[2][128 * LDS_STRIDE];
  __shared__ float wlds[128];

  const int tid  = threadIdx.x;
  const int lane = tid & 31, wave = tid >> 5;
  const int wm   = wave >> 2;          // 0..1 (M)
  const int wn   = wave & 3;           // 0..3 (N)
  const int half = lane >> 4;          // 0..1
  const int mrow = lane & 15;          // row (A) / column (B) within a tile
  const int m0 = blockIdx.y * 128, n0 = blockIdx.x * 128;

  if (MODE == 1 && tid < 128)
    wlds[tid] = wtg[(long long)(m0 + tid) * G_NUM + g];

  // staging coordinates: 512 16B chunks, 2 per thread
  const int r0 = (tid)       >> 2, c0 = ((tid)       & 3) << 3;
  const int r1 = (tid + 256) >> 2, c1 = ((tid + 256) & 3) << 3;
  const _Float16* aP0 = A  + (long long)(m0 + r0) * K + c0;
  const _Float16* aP1 = A  + (long long)(m0 + r1) * K + c1;
  const _Float16* g0P = B0 + (long long)(n0 + r0) * K + c0;
  const _Float16* g1P = B0 + (long long)(n0 + r1) * K + c1;
  const _Float16* u0P = B1 + (long long)(n0 + r0) * K + c0;
  const _Float16* u1P = B1 + (long long)(n0 + r1) * K + c1;
  const int lo0 = r0 * LDS_STRIDE + c0, lo1 = r1 * LDS_STRIDE + c1;

  v8f accG[4][2], accU[4][2];
#pragma unroll
  for (int i = 0; i < 4; ++i)
#pragma unroll
    for (int j = 0; j < 2; ++j)
#pragma unroll
      for (int e = 0; e < 8; ++e) { accG[i][j][e] = 0.f; accU[i][j][e] = 0.f; }

  auto issueTile = [&](int k0, int buf) {
    copy16_g2l(aP0 + k0, &lA [buf][lo0]);
    copy16_g2l(aP1 + k0, &lA [buf][lo1]);
    copy16_g2l(g0P + k0, &lB0[buf][lo0]);
    copy16_g2l(g1P + k0, &lB0[buf][lo1]);
    if (MODE == 0) {
      copy16_g2l(u0P + k0, &lB1[buf][lo0]);
      copy16_g2l(u1P + k0, &lB1[buf][lo1]);
    }
  };
  constexpr int NPEND = (MODE == 0) ? 6 : 4;  // asyncs per thread per tile

  issueTile(0, 0);                      // prologue

  const int nk = K / 32;
  for (int kk = 0; kk < nk; ++kk) {
    const int buf = kk & 1;
    if (kk + 1 < nk) {
      issueTile((kk + 1) * 32, buf ^ 1);  // next tile in flight
      WAIT_ASYNC(NPEND);                  // current tile resident
    } else {
      WAIT_ASYNC(0);
    }
    __syncthreads();                      // every wave waited on its own asyncs

    // ---- fragments (contiguous k order, same for A and B) ----
    // A (16x32 f16): lane(0..15)=row, elem i -> K = half*16 + i
    // B (32x16 f16): lane=column,      elem i -> K = half*16 + i
    const _Float16* la = lA[buf];
    const _Float16* l0 = lB0[buf];
    const _Float16* l1 = lB1[buf];
    union Fr { v16h v; v8h h[2]; } af[4], bg[2], bu[2];
#pragma unroll
    for (int mt = 0; mt < 4; ++mt) {
      int r = wm * 64 + mt * 16 + mrow;
      af[mt].h[0] = *(const v8h*)(&la[r * LDS_STRIDE + half * 16]);
      af[mt].h[1] = *(const v8h*)(&la[r * LDS_STRIDE + half * 16 + 8]);
    }
#pragma unroll
    for (int nt = 0; nt < 2; ++nt) {
      int r = wn * 32 + nt * 16 + mrow;
      bg[nt].h[0] = *(const v8h*)(&l0[r * LDS_STRIDE + half * 16]);
      bg[nt].h[1] = *(const v8h*)(&l0[r * LDS_STRIDE + half * 16 + 8]);
      if (MODE == 0) {
        bu[nt].h[0] = *(const v8h*)(&l1[r * LDS_STRIDE + half * 16]);
        bu[nt].h[1] = *(const v8h*)(&l1[r * LDS_STRIDE + half * 16 + 8]);
      }
    }

#pragma unroll
    for (int mt = 0; mt < 4; ++mt)
#pragma unroll
      for (int nt = 0; nt < 2; ++nt) {
        accG[mt][nt] = __builtin_amdgcn_wmma_f32_16x16x32_f16(
            false, af[mt].v, false, bg[nt].v, (short)0, accG[mt][nt],
            false, false);
        if (MODE == 0)
          accU[mt][nt] = __builtin_amdgcn_wmma_f32_16x16x32_f16(
              false, af[mt].v, false, bu[nt].v, (short)0, accU[mt][nt],
              false, false);
      }
    __syncthreads();                      // tile fully consumed
  }

  // ---- epilogue: C frag lane = column, VGPR v -> row half*8+v ----
#pragma unroll
  for (int mt = 0; mt < 4; ++mt) {
    int rb = wm * 64 + mt * 16 + half * 8;
    size_t rowBase = (size_t)(m0 + rb) * (size_t)N;
#pragma unroll
    for (int nt = 0; nt < 2; ++nt) {
      int col = n0 + wn * 32 + nt * 16 + mrow;
      if (MODE == 0) {
        _Float16* p = actOut + rowBase + col;
#pragma unroll
        for (int v = 0; v < 8; ++v) {
          float gv = accG[mt][nt][v];
          float uv = accU[mt][nt][v];
          float r = (gv / (1.f + __expf(-gv))) * uv;   // silu(g)*u
          *p = (_Float16)r;
          p += N;
        }
      } else {
        float* p = C + rowBase + col;
#pragma unroll
        for (int v = 0; v < 8; ++v) {
          float val = wlds[rb + v] * accG[mt][nt][v];
          *p = accum ? (*p + val) : val;
          p += N;
        }
      }
    }
  }
}

// ------------------------------- launch ------------------------------------
extern "C" void kernel_launch(void* const* d_in, const int* in_sizes, int n_in,
                              void* d_out, int out_size, void* d_ws,
                              size_t ws_size, hipStream_t stream) {
  (void)in_sizes; (void)n_in; (void)out_size; (void)ws_size;

  const float* x   = (const float*)d_in[0];  // [1,2048,2048]
  const float* gw  = (const float*)d_in[1];  // [32,2048]
  const float* gup = (const float*)d_in[2];  // [32,1536,2048]
  const float* dnp = (const float*)d_in[3];  // [32,2048,768]
  const int*   mg  = (const int*)d_in[4];    // [32]
  const int*   dom = (const int*)d_in[5];    // [8]
  float* out = (float*)d_out;                // [1,2048,2048]

  char* ws = (char*)d_ws;
  const size_t XH  = (size_t)T_TOK * H_DIM * 2;          // 8.4 MB x f16
  const size_t WGU = (size_t)(2 * I_DIM) * H_DIM * 2;    // 6.3 MB gu f16
  const size_t WDN = (size_t)H_DIM * I_DIM * 2;          // 3.1 MB dn f16
  const size_t AB  = (size_t)T_TOK * I_DIM * 2;          // 3.1 MB act f16
  _Float16* xh   = (_Float16*)ws;
  _Float16* wguf = (_Float16*)(ws + XH);
  _Float16* wdnf = (_Float16*)(ws + XH + WGU);
  _Float16* actb = (_Float16*)(ws + XH + WGU + WDN);
  float*    wtg  = (float*)(ws + XH + WGU + WDN + AB);   // 64 KB

  convert_x_kernel<<<(T_TOK * H_DIM) / (256 * 4), 256, 0, stream>>>(x, xh);
  router_kernel<<<T_TOK / 8, 256, 0, stream>>>(x, gw, mg, wtg);

  for (int g = 0; g < G_NUM; ++g) {
    // gather + convert dominant expert's weights to f16
    convert_w_kernel<<<(2 * I_DIM * H_DIM) / (256 * 8), 256, 0, stream>>>(
        gup, (long long)(2 * I_DIM) * H_DIM, dom, g, wguf);
    convert_w_kernel<<<(H_DIM * I_DIM) / (256 * 8), 256, 0, stream>>>(
        dnp, (long long)H_DIM * I_DIM, dom, g, wdnf);

    // act = silu(x*gate^T) * (x*up^T) : M=2048, N=768, K=2048 (dual-B fused)
    gemm_wmma_kernel<0><<<dim3(I_DIM / 128, T_TOK / 128), 256, 0, stream>>>(
        xh, wguf, wguf + (size_t)I_DIM * H_DIM, actb, nullptr, nullptr,
        I_DIM, H_DIM, g, 0);

    // out (+)= w_tg[:,g] * (act * down^T) : M=2048, N=2048, K=768
    gemm_wmma_kernel<1><<<dim3(H_DIM / 128, T_TOK / 128), 256, 0, stream>>>(
        actb, wdnf, nullptr, nullptr, out, wtg,
        H_DIM, I_DIM, g, g ? 1 : 0);
  }
}
